// SingleHeadOriginLayer_76819785056890
// MI455X (gfx1250) — matrix operand order repaired and checked by reading, not compiled, not run on hardware
//
#include <hip/hip_runtime.h>
#include <hip/hip_bf16.h>

// ---------------- WMMA fragment types (CDNA5 / gfx1250, wave32) -------------
typedef __attribute__((ext_vector_type(16))) _Float16 v16h;
typedef __attribute__((ext_vector_type(8)))  _Float16 v8h;
typedef __attribute__((ext_vector_type(8)))  float    v8f;

#define SLOPE 0.01f

__device__ __forceinline__ v8f wmma_f16(v16h a, v16h b, v8f c) {
    // D = A(16x32 f16) * B(32x16 f16) + C(16x16 f32)
    return __builtin_amdgcn_wmma_f32_16x16x32_f16(
        false, a, false, b, (short)0, c, false, false);
}

// A fragment: lane holds row M = lane%16; contiguous K run [16g..16g+15]
// `row` points at A_row_base + kb*32 (f16, row-major)
__device__ __forceinline__ v16h frag_a(const _Float16* row, int g) {
    v8h lo = *(const v8h*)(row + 16 * g);
    v8h hi = *(const v8h*)(row + 16 * g + 8);
    v16h a;
#pragma unroll
    for (int i = 0; i < 8; ++i) { a[i] = lo[i]; a[8 + i] = hi[i]; }
    return a;
}

// A fragment built from an f32 row (converted on the fly)
__device__ __forceinline__ v16h frag_a_f32(const float* row, int g) {
    v16h a;
    const float* p0 = row + 16 * g;
    const float* p1 = row + 16 * g + 8;
#pragma unroll
    for (int i = 0; i < 8; ++i) { a[i] = (_Float16)p0[i]; a[8 + i] = (_Float16)p1[i]; }
    return a;
}

// B fragment: lane holds column N = lane%16; contiguous K run [kb*32+16g .. +15]
// Bt is [Nout][K] f16 (pre-transposed weight)
__device__ __forceinline__ v16h frag_b(const _Float16* bt, int K, int nb, int kb, int lane) {
    int n = lane & 15, g = lane >> 4;
    return *(const v16h*)(bt + (size_t)(nb * 16 + n) * K + kb * 32 + g * 16);
}

// order-preserving float <-> uint for atomicMax-based segment max
__device__ __forceinline__ unsigned f2ord(float f) {
    unsigned u = __float_as_uint(f);
    return (u & 0x80000000u) ? ~u : (u | 0x80000000u);
}
__device__ __forceinline__ float ord2f(unsigned u) {
    return (u & 0x80000000u) ? __uint_as_float(u & 0x7fffffffu) : __uint_as_float(~u);
}

__device__ __forceinline__ float leaky(float v) { return v > 0.f ? v : SLOPE * v; }
__device__ __forceinline__ float sigm(float v) { return 1.f / (1.f + __expf(-v)); }

// ---------------- small utility kernels -------------------------------------
__global__ void k_fill_u32(unsigned* p, unsigned v, size_t n) {
    size_t i = (size_t)blockIdx.x * blockDim.x + threadIdx.x;
    if (i < n) p[i] = v;
}
__global__ void k_f32_to_f16(const float* __restrict__ w, _Float16* __restrict__ o, size_t n) {
    size_t i = (size_t)blockIdx.x * blockDim.x + threadIdx.x;
    if (i < n) o[i] = (_Float16)w[i];
}
// w [K][Nout] (in-major) -> wt [Nout][K] f16
__global__ void k_wt_transpose(const float* __restrict__ w, _Float16* __restrict__ wt,
                               int K, int Nout) {
    int i = blockIdx.x * blockDim.x + threadIdx.x;
    if (i >= K * Nout) return;
    int k = i / Nout, n = i % Nout;
    wt[(size_t)n * K + k] = (_Float16)w[i];
}
// out[i] = dot(A[i,0:64], v)
__global__ void k_node_dot(const float* __restrict__ A, const float* __restrict__ v,
                           float* __restrict__ out, int nrows) {
    int i = blockIdx.x * blockDim.x + threadIdx.x;
    if (i >= nrows) return;
    const float* r = A + (size_t)i * 64;
    float s = 0.f;
#pragma unroll
    for (int k = 0; k < 64; ++k) s += r[k] * v[k];
    out[i] = s;
}

// ---------------- generic node-level WMMA GEMM ------------------------------
// C[nrows,Nout] = act(A[nrows,K](f16) @ Bt[Nout,K]^T + bias)
// act: 0 none, 1 leaky_relu
__global__ __launch_bounds__(256) void k_node_gemm(
    const _Float16* __restrict__ A, const _Float16* __restrict__ Bt,
    const float* __restrict__ bias, float* __restrict__ Cf,
    _Float16* __restrict__ Ch, int nrows, int K, int Nout, int act) {
    int wave = (blockIdx.x * blockDim.x + threadIdx.x) >> 5;
    int lane = threadIdx.x & 31;
    int row0 = wave * 16;
    if (row0 >= nrows) return;               // wave-uniform: EXEC stays all-1
    int n = lane & 15, g = lane >> 4;
    int arow = row0 + n; if (arow >= nrows) arow = nrows - 1;
    const _Float16* Arow = A + (size_t)arow * K;
    int kblocks = K >> 5;
    for (int nb = 0; nb < (Nout >> 4); ++nb) {
        v8f c = {};
        for (int kb = 0; kb < kblocks; ++kb) {
            v16h a = frag_a(Arow + kb * 32, g);
            v16h b = frag_b(Bt, K, nb, kb, lane);
            c = wmma_f16(a, b, c);
        }
        float bv = bias ? bias[nb * 16 + n] : 0.f;
#pragma unroll
        for (int r = 0; r < 8; ++r) {
            int row = row0 + r + 8 * g;
            if (row < nrows) {
                float v = c[r] + bv;
                if (act == 1) v = leaky(v);
                size_t idx = (size_t)row * Nout + nb * 16 + n;
                if (Cf) Cf[idx] = v;
                if (Ch) Ch[idx] = (_Float16)v;
            }
        }
    }
}

// ---------------- GATEConv: attention logits --------------------------------
// he = leaky([x[src] | edge_attr] @ W1); logit = leaky(he.att_l + dr[dst])
__global__ __launch_bounds__(256) void k_gate_logits(
    const _Float16* __restrict__ xh, const float* __restrict__ ea,
    const _Float16* __restrict__ w1t, const float* __restrict__ att_l,
    const float* __restrict__ dr, const int* __restrict__ src,
    const int* __restrict__ dst, float* __restrict__ alpha,
    unsigned* __restrict__ nmax, int E) {
    int wave = (blockIdx.x * blockDim.x + threadIdx.x) >> 5;
    int lane = threadIdx.x & 31;
    int e0 = wave * 16;
    if (e0 >= E) return;
    int n = lane & 15, g = lane >> 4;
    int eA = e0 + n; if (eA >= E) eA = E - 1;
    const _Float16* xrow = xh + (size_t)src[eA] * 64;
    const float* earow = ea + (size_t)eA * 32;
    float partial[8];
#pragma unroll
    for (int r = 0; r < 8; ++r) partial[r] = 0.f;
    for (int nb = 0; nb < 4; ++nb) {
        v8f c = {};
#pragma unroll
        for (int kb = 0; kb < 2; ++kb) {           // K 0..63 from x[src]
            v16h a = frag_a(xrow + kb * 32, g);
            v16h b = frag_b(w1t, 96, nb, kb, lane);
            c = wmma_f16(a, b, c);
        }
        {                                          // K 64..95 from edge_attr
            v16h a = frag_a_f32(earow, g);
            v16h b = frag_b(w1t, 96, nb, 2, lane);
            c = wmma_f16(a, b, c);
        }
        float al = att_l[nb * 16 + n];
#pragma unroll
        for (int r = 0; r < 8; ++r) partial[r] += leaky(c[r]) * al;
    }
    // reduce over the 16 lanes of each half-wave (columns of one row set)
#pragma unroll
    for (int r = 0; r < 8; ++r) {
        float s = partial[r];
        s += __shfl_xor(s, 1, 32);
        s += __shfl_xor(s, 2, 32);
        s += __shfl_xor(s, 4, 32);
        s += __shfl_xor(s, 8, 32);
        partial[r] = s;
    }
#pragma unroll
    for (int r = 0; r < 8; ++r) {
        if (n == r) {
            int e = e0 + r + 8 * g;
            if (e < E) {
                int d = dst[e];
                float lg = leaky(partial[r] + dr[d]);
                alpha[e] = lg;
                atomicMax(nmax + d, f2ord(lg));
            }
        }
    }
}

// exp(logit - seg_max) and segment sum
__global__ void k_expsum(float* __restrict__ alpha, const unsigned* __restrict__ nmax,
                         float* __restrict__ nsum, const int* __restrict__ dst, int E) {
    int e = blockIdx.x * blockDim.x + threadIdx.x;
    if (e >= E) return;
    int d = dst[e];
    float ex = __expf(alpha[e] - ord2f(nmax[d]));
    alpha[e] = ex;
    atomicAdd(nsum + d, ex);
}

// ---------------- GATEConv: message GEMM + weighted scatter-add -------------
__global__ __launch_bounds__(256) void k_gate_msg(
    const _Float16* __restrict__ xh, const _Float16* __restrict__ w2t,
    const float* __restrict__ alpha, const float* __restrict__ nsum,
    const int* __restrict__ src, const int* __restrict__ dst,
    float* __restrict__ h, int E) {
    int wave = (blockIdx.x * blockDim.x + threadIdx.x) >> 5;
    int lane = threadIdx.x & 31;
    int e0 = wave * 16;
    if (e0 >= E) return;
    int n = lane & 15, g = lane >> 4;
    int eA = e0 + n; if (eA >= E) eA = E - 1;
    const _Float16* xrow = xh + (size_t)src[eA] * 64;
    float wrow[8]; int drow[8];
#pragma unroll
    for (int r = 0; r < 8; ++r) {
        int e = e0 + r + 8 * g; if (e >= E) e = E - 1;
        int d = dst[e];
        drow[r] = d;
        wrow[r] = alpha[e] / (nsum[d] + 1e-16f);
    }
    for (int nb = 0; nb < 4; ++nb) {
        v8f c = {};
#pragma unroll
        for (int kb = 0; kb < 2; ++kb) {
            v16h a = frag_a(xrow + kb * 32, g);
            v16h b = frag_b(w2t, 64, nb, kb, lane);
            c = wmma_f16(a, b, c);
        }
#pragma unroll
        for (int r = 0; r < 8; ++r) {
            int e = e0 + r + 8 * g;
            if (e < E)
                atomicAdd(h + (size_t)drow[r] * 64 + nb * 16 + n, c[r] * wrow[r]);
        }
    }
}

// ---------------- GATConv: logits / scatter ---------------------------------
__global__ void k_gat_logits(const float* __restrict__ ssrc, const float* __restrict__ sdst,
                             const int* __restrict__ src, const int* __restrict__ dst,
                             float* __restrict__ alpha, unsigned* __restrict__ nmax, int E) {
    int e = blockIdx.x * blockDim.x + threadIdx.x;
    if (e >= E) return;
    float lg = leaky(ssrc[src[e]] + sdst[dst[e]]);
    alpha[e] = lg;
    atomicMax(nmax + dst[e], f2ord(lg));
}

__global__ void k_gat_scatter(const float* __restrict__ xt, const float* __restrict__ alpha,
                              const float* __restrict__ nsum, const int* __restrict__ src,
                              const int* __restrict__ dst, float* __restrict__ h, int E) {
    size_t t = (size_t)blockIdx.x * blockDim.x + threadIdx.x;
    int e = (int)(t >> 4);
    if (e >= E) return;
    int c4 = (int)(t & 15) << 2;
    int s = src[e], d = dst[e];
    float w = alpha[e] / (nsum[d] + 1e-16f);
    const float* xr = xt + (size_t)s * 64 + c4;
    float* hr = h + (size_t)d * 64 + c4;
#pragma unroll
    for (int k = 0; k < 4; ++k) atomicAdd(hr + k, xr[k] * w);
}

// bias + activation (act 0 = elu, 1 = relu), also emit f16 copy
__global__ void k_bias_act(float* __restrict__ h, _Float16* __restrict__ hh,
                           const float* __restrict__ bias, int nrows, int act) {
    size_t i = (size_t)blockIdx.x * blockDim.x + threadIdx.x;
    if (i >= (size_t)nrows * 64) return;
    float v = h[i] + bias[i & 63];
    if (act) v = v > 0.f ? v : 0.f;
    else     v = v > 0.f ? v : (__expf(v) - 1.f);
    h[i] = v;
    hh[i] = (_Float16)v;
}

// GRUCell + relu, in-place on x (element (i,j) is fully independent)
__global__ void k_gru(const float* __restrict__ gi, const float* __restrict__ gh,
                      float* __restrict__ x, _Float16* __restrict__ xh, int nrows) {
    size_t i = (size_t)blockIdx.x * blockDim.x + threadIdx.x;
    if (i >= (size_t)nrows * 64) return;
    size_t row = i >> 6; int j = (int)(i & 63);
    const float* gir = gi + row * 192;
    const float* ghr = gh + row * 192;
    float r = sigm(gir[j] + ghr[j]);
    float z = sigm(gir[64 + j] + ghr[64 + j]);
    float nc = tanhf(gir[128 + j] + r * ghr[128 + j]);
    float xp = x[i];
    float v = (1.f - z) * nc + z * xp;
    v = v > 0.f ? v : 0.f;
    x[i] = v;
    xh[i] = (_Float16)v;
}

// global add pool: out[batch[i]] += nf[i]
__global__ void k_pool(const float* __restrict__ nf, const int* __restrict__ batch,
                       float* __restrict__ out, int nrows) {
    size_t t = (size_t)blockIdx.x * blockDim.x + threadIdx.x;
    int i = (int)(t >> 4);
    if (i >= nrows) return;
    int c4 = (int)(t & 15) << 2;
    int b = batch[i];
    const float* r = nf + (size_t)i * 64 + c4;
    float* o = out + (size_t)b * 64 + c4;
#pragma unroll
    for (int k = 0; k < 4; ++k) atomicAdd(o + k, r[k]);
}

// ---------------- host orchestration ----------------------------------------
static inline size_t cdiv(size_t a, size_t b) { return (a + b - 1) / b; }

extern "C" void kernel_launch(void* const* d_in, const int* in_sizes, int n_in,
                              void* d_out, int out_size, void* d_ws, size_t ws_size,
                              hipStream_t stream) {
    const float* x_in    = (const float*)d_in[0];
    const int*   eidx    = (const int*)d_in[1];
    const float* e_attr  = (const float*)d_in[2];
    const int*   batch   = (const int*)d_in[3];
    const float* lin1_w  = (const float*)d_in[4];
    const float* lin1_b  = (const float*)d_in[5];
    const float* g1_w    = (const float*)d_in[6];
    const float* g2_w    = (const float*)d_in[7];
    const float* att_l   = (const float*)d_in[8];
    const float* att_r   = (const float*)d_in[9];
    const float* gate_b  = (const float*)d_in[10];
    const float* g0_wih  = (const float*)d_in[11];
    const float* g0_whh  = (const float*)d_in[12];
    const float* g0_bih  = (const float*)d_in[13];
    const float* g0_bhh  = (const float*)d_in[14];
    const float* gat_w   = (const float*)d_in[15];
    const float* a_src   = (const float*)d_in[16];
    const float* a_dst   = (const float*)d_in[17];
    const float* gat_b   = (const float*)d_in[18];
    const float* gr_wih  = (const float*)d_in[19];
    const float* gr_whh  = (const float*)d_in[20];
    const float* gr_bih  = (const float*)d_in[21];
    const float* gr_bhh  = (const float*)d_in[22];
    const float* lin2_w  = (const float*)d_in[23];
    const float* lin2_b  = (const float*)d_in[24];

    const int N = in_sizes[0] / 64;
    const int E = in_sizes[1] / 2;
    const int* src = eidx;
    const int* dst = eidx + E;

    // workspace arena (256B aligned chunks)
    char* p = (char*)d_ws;
    auto alloc = [&](size_t bytes) {
        void* r = (void*)p;
        p += (bytes + 255) & ~((size_t)255);
        return r;
    };
    float*     xf    = (float*)alloc((size_t)N * 64 * 4);
    _Float16*  xh    = (_Float16*)alloc((size_t)N * 64 * 2);
    float*     xt    = (float*)alloc((size_t)N * 64 * 4);
    float*     hbuf  = (float*)alloc((size_t)N * 64 * 4);
    _Float16*  hh    = (_Float16*)alloc((size_t)N * 64 * 2);
    float*     gi    = (float*)alloc((size_t)N * 192 * 4);
    float*     gh    = (float*)alloc((size_t)N * 192 * 4);
    float*     alpha = (float*)alloc((size_t)E * 4);
    unsigned*  nmax  = (unsigned*)alloc((size_t)N * 4);
    float*     nsum  = (float*)alloc((size_t)N * 4);
    float*     dr    = (float*)alloc((size_t)N * 4);
    float*     dsr   = (float*)alloc((size_t)N * 4);
    _Float16*  lin1t = (_Float16*)alloc(64 * 64 * 2);
    _Float16*  g1t   = (_Float16*)alloc(64 * 96 * 2);
    _Float16*  g2t   = (_Float16*)alloc(64 * 64 * 2);
    _Float16*  w0ih  = (_Float16*)alloc(192 * 64 * 2);
    _Float16*  w0hh  = (_Float16*)alloc(192 * 64 * 2);
    _Float16*  gatT  = (_Float16*)alloc(2 * 64 * 64 * 2);
    _Float16*  wih   = (_Float16*)alloc(2 * 192 * 64 * 2);
    _Float16*  whh   = (_Float16*)alloc(2 * 192 * 64 * 2);
    _Float16*  lin2t = (_Float16*)alloc(64 * 64 * 2);

    const int TPB = 256;
    auto g1d = [&](size_t n) { return dim3((unsigned)cdiv(n, TPB)); };
    size_t nodeWaves = cdiv(N, 16);
    dim3 gridNode((unsigned)cdiv(nodeWaves * 32, TPB));
    size_t edgeWaves = cdiv(E, 16);
    dim3 gridEdgeW((unsigned)cdiv(edgeWaves * 32, TPB));

    // ---- weight conversion (f32 -> f16, transposed to [Nout][K]) ----
    k_wt_transpose<<<g1d(64 * 64), TPB, 0, stream>>>(lin1_w, lin1t, 64, 64);
    k_wt_transpose<<<g1d(96 * 64), TPB, 0, stream>>>(g1_w, g1t, 96, 64);
    k_wt_transpose<<<g1d(64 * 64), TPB, 0, stream>>>(g2_w, g2t, 64, 64);
    k_f32_to_f16<<<g1d(192 * 64), TPB, 0, stream>>>(g0_wih, w0ih, 192 * 64);
    k_f32_to_f16<<<g1d(192 * 64), TPB, 0, stream>>>(g0_whh, w0hh, 192 * 64);
    for (int l = 0; l < 2; ++l)
        k_wt_transpose<<<g1d(64 * 64), TPB, 0, stream>>>(gat_w + l * 4096, gatT + l * 4096, 64, 64);
    k_f32_to_f16<<<g1d(2 * 192 * 64), TPB, 0, stream>>>(gr_wih, wih, 2 * 192 * 64);
    k_f32_to_f16<<<g1d(2 * 192 * 64), TPB, 0, stream>>>(gr_whh, whh, 2 * 192 * 64);
    k_wt_transpose<<<g1d(64 * 64), TPB, 0, stream>>>(lin2_w, lin2t, 64, 64);

    // ---- lin1 + leaky_relu ----
    k_f32_to_f16<<<g1d((size_t)N * 64), TPB, 0, stream>>>(x_in, hh, (size_t)N * 64);
    k_node_gemm<<<gridNode, TPB, 0, stream>>>(hh, lin1t, lin1_b, xf, xh, N, 64, 64, 1);

    // ---- GATEConv ----
    k_node_dot<<<g1d(N), TPB, 0, stream>>>(xf, att_r, dr, N);
    k_fill_u32<<<g1d(N), TPB, 0, stream>>>(nmax, 0u, N);
    k_fill_u32<<<g1d(N), TPB, 0, stream>>>((unsigned*)nsum, 0u, N);
    k_fill_u32<<<g1d((size_t)N * 64), TPB, 0, stream>>>((unsigned*)hbuf, 0u, (size_t)N * 64);
    k_gate_logits<<<gridEdgeW, TPB, 0, stream>>>(xh, e_attr, g1t, att_l, dr, src, dst, alpha, nmax, E);
    k_expsum<<<g1d(E), TPB, 0, stream>>>(alpha, nmax, nsum, dst, E);
    k_gate_msg<<<gridEdgeW, TPB, 0, stream>>>(xh, g2t, alpha, nsum, src, dst, hbuf, E);
    k_bias_act<<<g1d((size_t)N * 64), TPB, 0, stream>>>(hbuf, hh, gate_b, N, 0 /*elu*/);
    // GRU0
    k_node_gemm<<<gridNode, TPB, 0, stream>>>(hh, w0ih, g0_bih, gi, nullptr, N, 64, 192, 0);
    k_node_gemm<<<gridNode, TPB, 0, stream>>>(xh, w0hh, g0_bhh, gh, nullptr, N, 64, 192, 0);
    k_gru<<<g1d((size_t)N * 64), TPB, 0, stream>>>(gi, gh, xf, xh, N);

    // ---- 2 x (GATConv + GRU) ----
    for (int l = 0; l < 2; ++l) {
        k_node_gemm<<<gridNode, TPB, 0, stream>>>(xh, gatT + l * 4096, nullptr, xt, nullptr, N, 64, 64, 0);
        k_node_dot<<<g1d(N), TPB, 0, stream>>>(xt, a_src + l * 64, dsr, N);
        k_node_dot<<<g1d(N), TPB, 0, stream>>>(xt, a_dst + l * 64, dr, N);
        k_fill_u32<<<g1d(N), TPB, 0, stream>>>(nmax, 0u, N);
        k_fill_u32<<<g1d(N), TPB, 0, stream>>>((unsigned*)nsum, 0u, N);
        k_fill_u32<<<g1d((size_t)N * 64), TPB, 0, stream>>>((unsigned*)hbuf, 0u, (size_t)N * 64);
        k_gat_logits<<<g1d(E), TPB, 0, stream>>>(dsr, dr, src, dst, alpha, nmax, E);
        k_expsum<<<g1d(E), TPB, 0, stream>>>(alpha, nmax, nsum, dst, E);
        k_gat_scatter<<<g1d((size_t)E * 16), TPB, 0, stream>>>(xt, alpha, nsum, src, dst, hbuf, E);
        k_bias_act<<<g1d((size_t)N * 64), TPB, 0, stream>>>(hbuf, hh, gat_b + l * 64, N, 1 /*relu*/);
        k_node_gemm<<<gridNode, TPB, 0, stream>>>(hh, wih + l * 192 * 64, gr_bih + l * 192, gi, nullptr, N, 64, 192, 0);
        k_node_gemm<<<gridNode, TPB, 0, stream>>>(xh, whh + l * 192 * 64, gr_bhh + l * 192, gh, nullptr, N, 64, 192, 0);
        k_gru<<<g1d((size_t)N * 64), TPB, 0, stream>>>(gi, gh, xf, xh, N);
    }

    // ---- lin2 + global add pool ----
    k_node_gemm<<<gridNode, TPB, 0, stream>>>(xh, lin2t, lin2_b, xt, nullptr, N, 64, 64, 0);
    k_fill_u32<<<g1d((size_t)out_size), TPB, 0, stream>>>((unsigned*)d_out, 0u, (size_t)out_size);
    k_pool<<<g1d((size_t)N * 16), TPB, 0, stream>>>(xt, batch, (float*)d_out, N);
}
